// Int8Linear_47854525612254
// MI455X (gfx1250) — compile-verified, matrix operand order checked by reading, not run on hardware
//
#include <hip/hip_runtime.h>

// ---- vector types ----------------------------------------------------------
typedef __attribute__((ext_vector_type(16))) _Float16    v16h;
typedef __attribute__((ext_vector_type(8)))  _Float16    v8h;
typedef __attribute__((ext_vector_type(4)))  _Float16    v4h;
typedef __attribute__((ext_vector_type(8)))  float       v8f;
typedef __attribute__((ext_vector_type(4)))  float       f32x4;
typedef __attribute__((ext_vector_type(16))) signed char i8x16;

// ---- problem constants -----------------------------------------------------
#define K_DIM 4096
#define N_DIM 11008
#define BM 128
#define BN 128
#define BK 64
#define LDA 72   // halves per LDS row (64 + 8 pad -> 144B stride, conflict-free)
#define LDB 72

__global__ __launch_bounds__(256) void int8_linear_wmma_kernel(
    const float* __restrict__ x,          // [M, K] fp32
    const signed char* __restrict__ w,    // [N, K] int8
    const float* __restrict__ scales,     // [N]
    const float* __restrict__ bias,       // [N]
    float* __restrict__ out)              // [M, N] fp32
{
    __shared__ _Float16 sA[BM * LDA];     // 18432 B
    __shared__ _Float16 sB[BN * LDB];     // 18432 B

    const int tid  = threadIdx.x;
    const int lane = tid & 31;            // wave32
    const int wave = tid >> 5;            // 8 waves
    const int wm   = wave & 1;            // 0..1 -> 64-row slab
    const int wn   = wave >> 1;           // 0..3 -> 32-col slab
    const int l16  = lane & 15;
    const int lhi  = lane >> 4;           // 0 or 1 (half-wave)

    const int bn0 = blockIdx.x * BN;
    const int bm0 = blockIdx.y * BM;

    v8f acc[4][2];
#pragma unroll
    for (int i = 0; i < 4; ++i)
#pragma unroll
        for (int j = 0; j < 2; ++j) {
            v8f z = {0.f, 0.f, 0.f, 0.f, 0.f, 0.f, 0.f, 0.f};
            acc[i][j] = z;
        }

    for (int k0 = 0; k0 < K_DIM; k0 += BK) {
        // ---- prefetch next K-tile into cache hierarchy (global_prefetch_b8)
        if (k0 + BK < K_DIM) {
            __builtin_prefetch(x + (size_t)(bm0 + (tid >> 1)) * K_DIM + k0 + BK, 0, 0);
            __builtin_prefetch(w + (size_t)(bn0 + (tid >> 1)) * K_DIM + k0 + BK, 0, 0);
        }

        // ---- stage A: 128 x 64 fp32 -> f16 in LDS (8 x float4 per thread)
#pragma unroll
        for (int i = 0; i < 8; ++i) {
            const int f   = tid + i * 256;      // 0..2047 float4-chunks
            const int row = f >> 4;             // 16 chunks per row
            const int c4  = (f & 15) << 2;      // column (floats)
            f32x4 v = *(const f32x4*)(x + (size_t)(bm0 + row) * K_DIM + k0 + c4);
            v4h h;
            h[0] = (_Float16)v[0]; h[1] = (_Float16)v[1];
            h[2] = (_Float16)v[2]; h[3] = (_Float16)v[3];
            *(v4h*)(sA + row * LDA + c4) = h;
        }

        // ---- stage B: 128 x 64 int8 -> f16 in LDS (2 x 16B per thread)
#pragma unroll
        for (int i = 0; i < 2; ++i) {
            const int c   = tid + i * 256;      // 0..511 16B-chunks
            const int row = c >> 2;             // 4 chunks per row
            const int cb  = (c & 3) << 4;       // column (elements)
            i8x16 q = *(const i8x16*)(w + (size_t)(bn0 + row) * K_DIM + k0 + cb);
            v8h h0, h1;
#pragma unroll
            for (int e = 0; e < 8; ++e) {
                h0[e] = (_Float16)(short)q[e];
                h1[e] = (_Float16)(short)q[8 + e];
            }
            *(v8h*)(sB + row * LDB + cb)     = h0;
            *(v8h*)(sB + row * LDB + cb + 8) = h1;
        }

        __syncthreads();

        // ---- compute: 2 K-chunks of 32, 4x2 fragments -> 16 WMMAs per wave
#pragma unroll
        for (int kk = 0; kk < BK; kk += 32) {
            const int klo = kk + (lhi << 3);    // half-wave K offset per A/B layout

            v16h afrag[4];
#pragma unroll
            for (int i = 0; i < 4; ++i) {
                const _Float16* p = sA + (wm * 64 + i * 16 + l16) * LDA + klo;
                v8h lo = *(const v8h*)p;        // K = klo .. klo+7
                v8h hi = *(const v8h*)(p + 16); // K = klo+16 .. klo+23
                afrag[i] = __builtin_shufflevector(lo, hi,
                    0, 1, 2, 3, 4, 5, 6, 7, 8, 9, 10, 11, 12, 13, 14, 15);
            }

            v16h bfrag[2];
#pragma unroll
            for (int j = 0; j < 2; ++j) {
                const _Float16* p = sB + (wn * 32 + j * 16 + l16) * LDB + klo;
                v8h lo = *(const v8h*)p;
                v8h hi = *(const v8h*)(p + 16);
                bfrag[j] = __builtin_shufflevector(lo, hi,
                    0, 1, 2, 3, 4, 5, 6, 7, 8, 9, 10, 11, 12, 13, 14, 15);
            }

#pragma unroll
            for (int i = 0; i < 4; ++i)
#pragma unroll
                for (int j = 0; j < 2; ++j)
                    acc[i][j] = __builtin_amdgcn_wmma_f32_16x16x32_f16(
                        /*neg_a=*/false, afrag[i],
                        /*neg_b=*/false, bfrag[j],
                        /*c_mod=*/(short)0, acc[i][j],
                        /*reuse_a=*/false, /*reuse_b=*/false);
        }

        __syncthreads();
    }

    // ---- epilogue: y = acc * scales[n] + bias[n]
    // C/D layout: VGPR v -> (M = v | v+8 by half-wave), lane%16 -> N
#pragma unroll
    for (int j = 0; j < 2; ++j) {
        const int n  = bn0 + wn * 32 + j * 16 + l16;
        const float sc = scales[n];
        const float bi = bias[n];
#pragma unroll
        for (int i = 0; i < 4; ++i) {
            const int mbase = bm0 + wm * 64 + i * 16 + (lhi << 3);
#pragma unroll
            for (int v = 0; v < 8; ++v) {
                out[(size_t)(mbase + v) * N_DIM + n] = acc[i][j][v] * sc + bi;
            }
        }
    }
}

extern "C" void kernel_launch(void* const* d_in, const int* in_sizes, int n_in,
                              void* d_out, int out_size, void* d_ws, size_t ws_size,
                              hipStream_t stream) {
    const float*       x      = (const float*)d_in[0];
    const signed char* weight = (const signed char*)d_in[1];
    const float*       scales = (const float*)d_in[2];
    const float*       bias   = (const float*)d_in[3];
    float*             out    = (float*)d_out;

    const int M = in_sizes[0] / K_DIM;    // 4 * 2048 = 8192
    dim3 grid(N_DIM / BN, M / BM);        // (86, 64)
    dim3 block(256);                      // 8 wave32 waves
    int8_linear_wmma_kernel<<<grid, block, 0, stream>>>(x, weight, scales, bias, out);
}
